// Encoder_39384850104484
// MI455X (gfx1250) — matrix operand order repaired and checked by reading, not compiled, not run on hardware
//
#include <hip/hip_runtime.h>

// ---------------------------------------------------------------------------
// Transformer encoder block on gfx1250 (MI455X), bf16 WMMA + fp32 accumulate.
// B=8, S=2048, D=1024.
// GEMMs: LDS-staged, double-buffered K-slabs fed by GLOBAL_LOAD_ASYNC_TO_LDS
// (ASYNCcnt / s_wait_asynccnt); fragment loads hoisted so WMMAs issue
// back-to-back after a single dscnt drain per slab.
// ---------------------------------------------------------------------------

typedef unsigned short u16;
typedef unsigned int   u32;
typedef __attribute__((ext_vector_type(16))) __bf16 bf16x16;
typedef __attribute__((ext_vector_type(8)))  float  f32x8;
typedef __attribute__((ext_vector_type(4)))  int    v4i;

struct alignas(16) U128 { u32 x, y, z, w; };

__device__ __forceinline__ u16 f32_to_bf16(float f) {
  u32 u = __builtin_bit_cast(u32, f);
  if ((u & 0x7F800000u) != 0x7F800000u)          // finite: round-to-nearest-even
    u += 0x7FFFu + ((u >> 16) & 1u);
  return (u16)(u >> 16);
}

// --------------------------- async copy to LDS -----------------------------
#if __has_builtin(__builtin_amdgcn_global_load_async_to_lds_b128)
#define HAVE_ASYNC_LDS 1
typedef __attribute__((address_space(1))) v4i gv4i;   // global
typedef __attribute__((address_space(3))) v4i lv4i;   // LDS
__device__ __forceinline__ void copy16_to_lds(const u16* gsrc, u16* ldst) {
  u16* g = const_cast<u16*>(gsrc);
  __builtin_amdgcn_global_load_async_to_lds_b128((gv4i*)g, (lv4i*)ldst, 0, 0);
}
__device__ __forceinline__ void async_join() {
#if __has_builtin(__builtin_amdgcn_s_wait_asynccnt)
  __builtin_amdgcn_s_wait_asynccnt(0);
#else
  asm volatile("s_wait_asynccnt 0x0" ::: "memory");
#endif
}
#else
#define HAVE_ASYNC_LDS 0
__device__ __forceinline__ void copy16_to_lds(const u16* gsrc, u16* ldst) {
  *(U128*)ldst = *(const U128*)gsrc;             // sync fallback
}
__device__ __forceinline__ void async_join() {}
#endif

// --------------------------- WMMA fragment helpers -------------------------
__device__ __forceinline__ bf16x16 frag_from(const u16* p0, const u16* p1) {
  union { U128 q[2]; bf16x16 v; } u;
  u.q[0] = *(const U128*)p0;
  u.q[1] = *(const U128*)p1;
  return u.v;
}

// A fragment (16x32 bf16 tile, row-major, leading dim ld). ISA layout:
// lane<16 -> M=lane, K={0..7,16..23}; lane>=16 -> M=lane-16, K={8..15,24..31}.
__device__ __forceinline__ bf16x16 load_a(const u16* tile, int ld, int kbase, int lane) {
  int m   = lane & 15;
  int off = (lane & 16) ? 8 : 0;
  const u16* r = tile + (size_t)m * ld + kbase + off;
  return frag_from(r, r + 16);
}

// B fragment (32x16) from B^T stored row-major [N x K], ld=K. ISA layout:
// lane<16 -> N=lane, K={0..7,16..23}; lane>=16 -> N=lane-16, K={8..15,24..31}.
__device__ __forceinline__ bf16x16 load_bt(const u16* bt, int ld, int kbase, int lane) {
  int n   = lane & 15;
  int off = (lane & 16) ? 8 : 0;
  const u16* r = bt + (size_t)n * ld + kbase + off;
  return frag_from(r, r + 16);
}

__device__ __forceinline__ f32x8 wmma_bf16(bf16x16 a, bf16x16 b, f32x8 c) {
  return __builtin_amdgcn_wmma_f32_16x16x32_bf16(false, a, false, b, (short)0, c,
                                                 false, false);
}

// --------------------------- staged GEMM core ------------------------------
// Block = 256 threads (8 waves), output tile 64(M) x 128(N).
// K consumed in 64-deep slabs staged in LDS, double buffered (55296 B).
#define SLAB_ELEMS (64 * 72 + 128 * 72)

__device__ __forceinline__ void stage_slab(const u16* __restrict__ Ablk, int lda,
                                           const u16* __restrict__ Bblk, int ldb,
                                           int k0, u16* dst, int tid) {
  u16* dA = dst;                 // 64 rows x 64 cols (stride 72)
  u16* dB = dst + 64 * 72;       // 128 rows x 64 cols (stride 72)
  for (int i = tid; i < 512; i += 256) {       // 64*64/8 chunks of 8 elems
    int r = i >> 3, c = (i & 7) * 8;
    copy16_to_lds(Ablk + (size_t)r * lda + k0 + c, dA + r * 72 + c);
  }
  for (int i = tid; i < 1024; i += 256) {      // 128*64/8 chunks
    int r = i >> 3, c = (i & 7) * 8;
    copy16_to_lds(Bblk + (size_t)r * ldb + k0 + c, dB + r * 72 + c);
  }
}

// Load ALL fragments for the slab first (20 x ds_load_b128), then issue the
// 8 WMMAs back-to-back: a single dscnt drain per slab instead of one per MMA.
__device__ __forceinline__ void compute_slab(const u16* buf, f32x8 acc[4],
                                             int lane, int wave) {
  const u16* bA = buf + (size_t)(wave >> 1) * 16 * 72;
  const u16* bB = buf + 64 * 72 + (size_t)(wave & 1) * 64 * 72;
  bf16x16 afr[2], bfr[2][4];
#pragma unroll
  for (int kk = 0; kk < 2; kk++) {
    afr[kk] = load_a(bA, 72, kk * 32, lane);
#pragma unroll
    for (int t = 0; t < 4; t++)
      bfr[kk][t] = load_bt(bB + (size_t)t * 16 * 72, 72, kk * 32, lane);
  }
#pragma unroll
  for (int kk = 0; kk < 2; kk++)
#pragma unroll
    for (int t = 0; t < 4; t++)
      acc[t] = wmma_bf16(afr[kk], bfr[kk][t], acc[t]);
}

__device__ __forceinline__ void gemm_staged(const u16* __restrict__ Ablk, int lda,
                                            const u16* __restrict__ Bblk, int ldb,
                                            int K, u16 (*smem)[SLAB_ELEMS],
                                            f32x8 acc[4]) {
  int tid = threadIdx.x, lane = tid & 31, wave = tid >> 5;
#pragma unroll
  for (int t = 0; t < 4; t++) acc[t] = {};
  int ns = K >> 6;
  stage_slab(Ablk, lda, Bblk, ldb, 0, smem[0], tid);
  async_join();
  __syncthreads();
  for (int s = 0; s < ns; s++) {
    if (s + 1 < ns)   // prefetch next slab while WMMAs consume current one
      stage_slab(Ablk, lda, Bblk, ldb, (s + 1) << 6, smem[(s + 1) & 1], tid);
    compute_slab(smem[s & 1], acc, lane, wave);
    async_join();
    __syncthreads();
  }
}

// ---------------------------------------------------------------------------
// Kernel 0: fp32 -> bf16 convert + transpose  (Wt[n*K + k] = W[k*N + n])
// ---------------------------------------------------------------------------
__global__ void conv_transpose_kernel(const float* __restrict__ W,
                                      u16* __restrict__ Wt, int K, int N) {
  long i = (long)blockIdx.x * blockDim.x + threadIdx.x;
  long total = (long)K * N;
  if (i >= total) return;
  int n = (int)(i / K), k = (int)(i % K);
  Wt[i] = f32_to_bf16(W[(size_t)k * N + n]);
}

// ---------------------------------------------------------------------------
// LayerNorm over D=1024, one 128-thread block per row, bf16 output.
// ---------------------------------------------------------------------------
__global__ void ln_bf16_kernel(const float* __restrict__ x, const float* __restrict__ w,
                               const float* __restrict__ b, u16* __restrict__ out) {
  __shared__ float red[128];
  int row = blockIdx.x;
  const float* xr = x + (size_t)row * 1024;
  float s = 0.f, s2 = 0.f;
  for (int i = threadIdx.x; i < 1024; i += 128) { float v = xr[i]; s += v; s2 += v * v; }
  red[threadIdx.x] = s; __syncthreads();
  for (int o = 64; o > 0; o >>= 1) {
    if ((int)threadIdx.x < o) red[threadIdx.x] += red[threadIdx.x + o];
    __syncthreads();
  }
  float mean = red[0] * (1.f / 1024.f);
  __syncthreads();
  red[threadIdx.x] = s2; __syncthreads();
  for (int o = 64; o > 0; o >>= 1) {
    if ((int)threadIdx.x < o) red[threadIdx.x] += red[threadIdx.x + o];
    __syncthreads();
  }
  float var = red[0] * (1.f / 1024.f) - mean * mean;
  float inv = rsqrtf(var + 1e-5f);
  u16* orow = out + (size_t)row * 1024;
  for (int i = threadIdx.x; i < 1024; i += 128)
    orow[i] = f32_to_bf16((xr[i] - mean) * inv * w[i] + b[i]);
}

// ---------------------------------------------------------------------------
// GEMM 1: qkv = hA[16384,1024] @ W_attn + b_attn  (K=1024, N=3072)
// Epilogue: q,k row-major bf16; v stored TRANSPOSED as vT[B][D][S] bf16.
// ---------------------------------------------------------------------------
__global__ void gemm_qkv_kernel(const u16* __restrict__ A, const u16* __restrict__ Bt,
                                const float* __restrict__ bias,
                                u16* __restrict__ qB, u16* __restrict__ kB,
                                u16* __restrict__ vT) {
  __shared__ alignas(16) u16 smem[2][SLAB_ELEMS];
  int lane = threadIdx.x & 31, wave = threadIdx.x >> 5;
  f32x8 acc[4];
  gemm_staged(A + (size_t)blockIdx.x * 64 * 1024, 1024,
              Bt + (size_t)blockIdx.y * 128 * 1024, 1024, 1024, smem, acc);

  int row0 = blockIdx.x * 64 + (wave >> 1) * 16;
  int col0 = blockIdx.y * 128 + (wave & 1) * 64;
  int mo = (lane & 16) ? 8 : 0, n = lane & 15;
#pragma unroll
  for (int t = 0; t < 4; t++) {
    int c = col0 + t * 16 + n;
    float bv = bias[c];
#pragma unroll
    for (int r = 0; r < 8; r++) {
      int s = row0 + r + mo;
      u16 h = f32_to_bf16(acc[t][r] + bv);
      if (c < 1024)            qB[(size_t)s * 1024 + c] = h;
      else if (c < 2048)       kB[(size_t)s * 1024 + (c - 1024)] = h;
      else {
        int bb = s >> 11, si = s & 2047, d = c - 2048;
        vT[((size_t)bb * 1024 + d) * 2048 + si] = h;
      }
    }
  }
}

// ---------------------------------------------------------------------------
// Flash-style single-head attention + residual. 1 block (8 waves) per
// (batch, 16-row block); online softmax over 128-key chunks.
// x2 = x + softmax(q k^T / 32) @ v
// ---------------------------------------------------------------------------
__global__ void attn_kernel(const u16* __restrict__ qB, const u16* __restrict__ kB,
                            const u16* __restrict__ vT, const float* __restrict__ x,
                            float* __restrict__ x2) {
  __shared__ alignas(16) u16  sQ[16 * 1024];      // q row block, bf16 (32 KB)
  __shared__ float            sS[16][132];        // score chunk fp32
  __shared__ alignas(16) u16  sP[16][136];        // exp probs, bf16
  __shared__ float            sM[16], sL[16], sC[16];

  int lane = threadIdx.x & 31, wave = threadIdx.x >> 5;
  int blk = blockIdx.x;
  int b  = blk >> 7;
  int r0 = (blk & 127) * 16;
  const u16* qrow = qB + ((size_t)b * 2048 + r0) * 1024;
  const u16* kbat = kB + (size_t)b * 2048 * 1024;
  const u16* vbat = vT + (size_t)b * 1024 * 2048;

  // Stage q block into LDS cooperatively (async path).
  for (int i = threadIdx.x; i < 16 * 1024 / 8; i += 256)
    copy16_to_lds(qrow + i * 8, sQ + i * 8);
  if (threadIdx.x < 16) { sM[threadIdx.x] = -1e30f; sL[threadIdx.x] = 0.f; }
  async_join();
  __syncthreads();

  f32x8 acc[8];
#pragma unroll
  for (int t = 0; t < 8; t++) acc[t] = {};
  int mo = (lane & 16) ? 8 : 0, n = lane & 15;

  for (int j = 0; j < 2048; j += 128) {
    // --- scores tile (16x16 per wave), software-pipelined fragment loads
    f32x8 sc = {};
    const u16* krow = kbat + (size_t)(j + wave * 16) * 1024;
    bf16x16 a0 = load_a(sQ, 1024, 0, lane);
    bf16x16 b0 = load_bt(krow, 1024, 0, lane);
    for (int k = 0; k < 1024; k += 32) {
      int kn = (k + 32 < 1024) ? (k + 32) : k;
      bf16x16 a1 = load_a(sQ, 1024, kn, lane);
      bf16x16 b1 = load_bt(krow, 1024, kn, lane);
      sc = wmma_bf16(a0, b0, sc);
      a0 = a1; b0 = b1;
    }
#pragma unroll
    for (int r = 0; r < 8; r++) sS[r + mo][wave * 16 + n] = sc[r] * 0.03125f;
    __syncthreads();

    // --- online softmax bookkeeping (one thread per row)
    if (threadIdx.x < 16) {
      int row = threadIdx.x;
      float mOld = sM[row], mNew = mOld;
      for (int t = 0; t < 128; t++) mNew = fmaxf(mNew, sS[row][t]);
      float cf = __expf(mOld - mNew);
      float sum = 0.f;
      for (int t = 0; t < 128; t++) {
        float p = __expf(sS[row][t] - mNew);
        sum += p;
        sP[row][t] = f32_to_bf16(p);
      }
      sM[row] = mNew;
      sL[row] = sL[row] * cf + sum;
      sC[row] = cf;
    }
    __syncthreads();

    // --- rescale accumulators, then acc += P(16x128) @ V slice.
#pragma unroll
    for (int t = 0; t < 8; t++)
#pragma unroll
      for (int r = 0; r < 8; r++) acc[t][r] *= sC[r + mo];

    // Per 32-deep k-step: load A frag + all 8 V frags, then 8 WMMAs.
#pragma unroll
    for (int kk = 0; kk < 128; kk += 32) {
      bf16x16 af = load_a(&sP[0][0], 136, kk, lane);
      bf16x16 bfr[8];
#pragma unroll
      for (int t = 0; t < 8; t++)
        bfr[t] = load_bt(vbat + (size_t)(wave * 128 + t * 16) * 2048, 2048,
                         j + kk, lane);
#pragma unroll
      for (int t = 0; t < 8; t++)
        acc[t] = wmma_bf16(af, bfr[t], acc[t]);
    }
    __syncthreads();
  }

  // Epilogue: residual add, divide by softmax denominator.
#pragma unroll
  for (int t = 0; t < 8; t++) {
#pragma unroll
    for (int r = 0; r < 8; r++) {
      int row = r + mo;
      size_t sg = (size_t)b * 2048 + r0 + row;
      int d = wave * 128 + t * 16 + n;
      x2[sg * 1024 + d] = x[sg * 1024 + d] + acc[t][r] / sL[row];
    }
  }
}

// ---------------------------------------------------------------------------
// GEMM 2: hC = gelu(hB[16384,1024] @ W_fc + b_fc)   (K=1024, N=4096, bf16 out)
// ---------------------------------------------------------------------------
__global__ void gemm_gelu_kernel(const u16* __restrict__ A, const u16* __restrict__ Bt,
                                 const float* __restrict__ bias, u16* __restrict__ out) {
  __shared__ alignas(16) u16 smem[2][SLAB_ELEMS];
  int lane = threadIdx.x & 31, wave = threadIdx.x >> 5;
  f32x8 acc[4];
  gemm_staged(A + (size_t)blockIdx.x * 64 * 1024, 1024,
              Bt + (size_t)blockIdx.y * 128 * 1024, 1024, 1024, smem, acc);

  int row0 = blockIdx.x * 64 + (wave >> 1) * 16;
  int col0 = blockIdx.y * 128 + (wave & 1) * 64;
  int mo = (lane & 16) ? 8 : 0, n = lane & 15;
#pragma unroll
  for (int t = 0; t < 4; t++) {
    int c = col0 + t * 16 + n;
    float bv = bias[c];
#pragma unroll
    for (int r = 0; r < 8; r++) {
      int s = row0 + r + mo;
      float v = acc[t][r] + bv;
      v = 0.5f * v * (1.f + erff(v * 0.70710678118654752f));   // exact (erf) GELU
      out[(size_t)s * 4096 + c] = f32_to_bf16(v);
    }
  }
}

// ---------------------------------------------------------------------------
// GEMM 3: out = hC[16384,4096] @ W_proj + b_proj   (K=4096, N=1024, fp32 out)
// ---------------------------------------------------------------------------
__global__ void gemm_out_kernel(const u16* __restrict__ A, const u16* __restrict__ Bt,
                                const float* __restrict__ bias, float* __restrict__ out) {
  __shared__ alignas(16) u16 smem[2][SLAB_ELEMS];
  int lane = threadIdx.x & 31, wave = threadIdx.x >> 5;
  f32x8 acc[4];
  gemm_staged(A + (size_t)blockIdx.x * 64 * 4096, 4096,
              Bt + (size_t)blockIdx.y * 128 * 4096, 4096, 4096, smem, acc);

  int row0 = blockIdx.x * 64 + (wave >> 1) * 16;
  int col0 = blockIdx.y * 128 + (wave & 1) * 64;
  int mo = (lane & 16) ? 8 : 0, n = lane & 15;
#pragma unroll
  for (int t = 0; t < 4; t++) {
    int c = col0 + t * 16 + n;
    float bv = bias[c];
#pragma unroll
    for (int r = 0; r < 8; r++) {
      int s = row0 + r + mo;
      out[(size_t)s * 1024 + c] = acc[t][r] + bv;
    }
  }
}

// ---------------------------------------------------------------------------
// Host side
// ---------------------------------------------------------------------------
extern "C" void kernel_launch(void* const* d_in, const int* in_sizes, int n_in,
                              void* d_out, int out_size, void* d_ws, size_t ws_size,
                              hipStream_t stream) {
  const float* x      = (const float*)d_in[0];
  const float* ln1_w  = (const float*)d_in[1];
  const float* ln1_b  = (const float*)d_in[2];
  const float* W_attn = (const float*)d_in[3];
  const float* b_attn = (const float*)d_in[4];
  const float* ln2_w  = (const float*)d_in[5];
  const float* ln2_b  = (const float*)d_in[6];
  const float* W_fc   = (const float*)d_in[7];
  const float* b_fc   = (const float*)d_in[8];
  const float* W_proj = (const float*)d_in[9];
  const float* b_proj = (const float*)d_in[10];
  float* out = (float*)d_out;

  // Workspace layout (bytes). Required: ~224.4 MB.
  char* ws = (char*)d_ws;
  u16*   wtA = (u16*)(ws);                         // [3072,1024] bf16
  u16*   wtF = (u16*)(ws + 6291456);               // [4096,1024] bf16
  u16*   wtP = (u16*)(ws + 14680064);              // [1024,4096] bf16
  u16*   hA  = (u16*)(ws + 23068672);              // [16384,1024] bf16 (reused as hB)
  u16*   qB  = (u16*)(ws + 56623104);              // [16384,1024] bf16
  u16*   kB  = (u16*)(ws + 90177536);              // [16384,1024] bf16
  u16*   vT  = (u16*)(ws + 123731968);             // [8,1024,2048] bf16
  float* x2  = (float*)(ws + 157286400);           // [16384,1024] f32
  u16*   hB  = hA;                                 // alias: hA dead after GEMM1
  u16*   hC  = qB;                                 // alias: q/k/vT dead after LN2
  const size_t required = 224395264;
  if (ws_size < required) return;

  conv_transpose_kernel<<<(1024 * 3072 + 255) / 256, 256, 0, stream>>>(W_attn, wtA, 1024, 3072);
  conv_transpose_kernel<<<(1024 * 4096 + 255) / 256, 256, 0, stream>>>(W_fc,   wtF, 1024, 4096);
  conv_transpose_kernel<<<(4096 * 1024 + 255) / 256, 256, 0, stream>>>(W_proj, wtP, 4096, 1024);

  ln_bf16_kernel<<<16384, 128, 0, stream>>>(x, ln1_w, ln1_b, hA);
  gemm_qkv_kernel<<<dim3(256, 24), 256, 0, stream>>>(hA, wtA, b_attn, qB, kB, vT);
  attn_kernel<<<1024, 256, 0, stream>>>(qB, kB, vT, x, x2);
  ln_bf16_kernel<<<16384, 128, 0, stream>>>(x2, ln2_w, ln2_b, hB);
  gemm_gelu_kernel<<<dim3(256, 32), 256, 0, stream>>>(hB, wtF, b_fc, hC);
  gemm_out_kernel<<<dim3(256, 8), 256, 0, stream>>>(hC, wtP, b_proj, out);
}
